// LocalAttention_42253888258442
// MI455X (gfx1250) — matrix operand order repaired and checked by reading, not compiled, not run on hardware
//
#include <hip/hip_runtime.h>

// ---------------- WMMA types (gfx1250, wave32) ----------------
typedef __attribute__((ext_vector_type(16))) __bf16 v16bf;
typedef __attribute__((ext_vector_type(8)))  float  v8f;
typedef __attribute__((ext_vector_type(4)))  unsigned int v4u;

union Frag {
  v16bf bf;
  v4u   q[2];
  unsigned int u[8];
};

__device__ __forceinline__ unsigned short f2bf(float f) {
  unsigned int b = __float_as_uint(f);
  b += 0x7FFFu + ((b >> 16) & 1u);   // round-to-nearest-even
  return (unsigned short)(b >> 16);
}

__device__ __forceinline__ v8f wmma_bf16(const Frag& a, const Frag& b, v8f c) {
  return __builtin_amdgcn_wmma_f32_16x16x32_bf16(false, a.bf, false, b.bf,
                                                 (short)0, c, false, false);
}

#define BATCH 8
#define CCH   256
#define HWN   1024
#define NHEAD 4
#define HD    64
#define NI    8
#define OQKV  768   // q(0..255) k(256..511) v(512..767)

// ---------------- kernel 0: zero the output ----------------
__global__ void la_zero_kernel(float* p, int n) {
  int i = blockIdx.x * blockDim.x + threadIdx.x;
  if (i < n) p[i] = 0.0f;
}

// ---------------- kernel 1: pack W_{q,k,v} -> bf16 [o][c], bias[768] ----------------
__global__ void la_prep_kernel(const float* wq, const float* wk, const float* wv,
                               const float* bq, const float* bk, const float* bv,
                               unsigned short* wqkv, float* biasqkv) {
  int o = blockIdx.x;          // 0..767
  int c = threadIdx.x;         // 0..255
  const float* src = (o < 256) ? (wq + (size_t)o * 256)
                   : (o < 512) ? (wk + (size_t)(o - 256) * 256)
                               : (wv + (size_t)(o - 512) * 256);
  wqkv[(size_t)o * 256 + c] = f2bf(src[c]);
  if (c == 0)
    biasqkv[o] = (o < 256) ? bq[o] : (o < 512) ? bk[o - 256] : bv[o - 512];
}

// ---------------- kernel 2: instance conv + softmax over I -> masks[I][B*HW] ----------------
__global__ void la_mask_kernel(const float* x, const float* w_inst,
                               const float* b_inst, float* masks) {
  int t = blockIdx.x * blockDim.x + threadIdx.x; // 0..8191  (b,n)
  int b = t >> 10, n = t & 1023;
  const float* xp = x + (size_t)b * CCH * HWN + n;
  float acc[NI];
#pragma unroll
  for (int i = 0; i < NI; ++i) acc[i] = b_inst[i];
  for (int c = 0; c < CCH; ++c) {
    float xv = xp[(size_t)c * HWN];
#pragma unroll
    for (int i = 0; i < NI; ++i) acc[i] += w_inst[i * CCH + c] * xv;
  }
  float mx = acc[0];
#pragma unroll
  for (int i = 1; i < NI; ++i) mx = fmaxf(mx, acc[i]);
  float s = 0.0f;
#pragma unroll
  for (int i = 0; i < NI; ++i) { acc[i] = __expf(acc[i] - mx); s += acc[i]; }
  float inv = 1.0f / s;
#pragma unroll
  for (int i = 0; i < NI; ++i) masks[(size_t)i * 8192 + t] = acc[i] * inv;
}

// ---------------- kernel 3: QKV projection GEMM (bf16 WMMA) ----------------
// out[b][o][n] = sum_c W[o][c] * (x[b][c][n] * mask[b][n]) + bias[o]
// grid (n_tiles=16, o_tiles=12, B=8), block 128 (4 waves); tile M=64, N=64, K-step 32
__global__ void la_qkv_kernel(const float* __restrict__ x, const float* __restrict__ mask,
                              const unsigned short* __restrict__ wqkv,
                              const float* __restrict__ biasqkv, float* __restrict__ qkv) {
  __shared__ unsigned short featT[64 * 32];  // [n_local][c_local] bf16
  int b  = blockIdx.z;
  int o0 = blockIdx.y * 64;
  int n0 = blockIdx.x * 64;
  int tid = threadIdx.x;
  int wave = tid >> 5, lane = tid & 31;
  int lm = lane & 15, hi = (lane >> 4) & 1;

  const float* xb = x + (size_t)b * CCH * HWN;
  const float* mb = mask + (size_t)b * HWN;

  v8f acc[4] = {};                 // wave: M=16 strip x N=64 (4 tiles)
  int orow = o0 + wave * 16 + lm;  // A row (same data both lane halves)
  int kbA  = hi * 8;               // A K sub-offset per lane half
  int kkB  = hi * 8;               // B K sub-offset per lane half

  int tn = tid & 63, tc = tid >> 6;
  for (int c0 = 0; c0 < CCH; c0 += 32) {
    __syncthreads();
    // stage feat tile, transposed to [n][c], bf16
    float mval = mb[n0 + tn];
#pragma unroll
    for (int cc = 0; cc < 16; ++cc) {
      int c = tc + 2 * cc;
      float xv = xb[(size_t)(c0 + c) * HWN + n0 + tn];
      featT[tn * 32 + c] = f2bf(xv * mval);
    }
    if (c0 + 32 < CCH)
      __builtin_prefetch(xb + (size_t)(c0 + 32) * HWN + n0 + tn, 0, 0);
    __syncthreads();

    Frag A;
    A.q[0] = *(const v4u*)(wqkv + (size_t)orow * 256 + c0 + kbA);        // K = kb..kb+7
    A.q[1] = *(const v4u*)(wqkv + (size_t)orow * 256 + c0 + 16 + kbA);   // K = 16+kb..
#pragma unroll
    for (int t = 0; t < 4; ++t) {
      Frag B;
      const unsigned short* bp = &featT[(t * 16 + lm) * 32 + kkB];
      B.q[0] = *(const v4u*)(bp);
      B.q[1] = *(const v4u*)(bp + 16);
      acc[t] = wmma_bf16(A, B, acc[t]);
    }
  }
  // epilogue: D layout -> qkv[b][o][n] + bias
#pragma unroll
  for (int t = 0; t < 4; ++t) {
#pragma unroll
    for (int r = 0; r < 8; ++r) {
      int oo = o0 + wave * 16 + r + 8 * hi;
      int nn = n0 + t * 16 + lm;
      qkv[((size_t)b * OQKV + oo) * HWN + nn] = acc[t][r] + biasqkv[oo];
    }
  }
}

// ---------------- kernel 4: l2-normalize q,k rows (over HW) + bf16 transpose packs ----
// q rows -> qnT[b][n][o] bf16 ; k rows -> knT[b][n][o] bf16 ; v rows -> vb[b][o][n] bf16
__global__ void la_norm_kernel(const float* __restrict__ qkv,
                               unsigned short* __restrict__ qnT,
                               unsigned short* __restrict__ knT,
                               unsigned short* __restrict__ vb) {
  __shared__ float red[8];
  __shared__ float stot;
  int bo = blockIdx.x;
  int b = bo / OQKV, o = bo % OQKV;
  const float* row = qkv + ((size_t)b * OQKV + o) * HWN;
  int t = threadIdx.x; // 256
  float v[4];
#pragma unroll
  for (int k = 0; k < 4; ++k) v[k] = row[t + k * 256];
  if (o < 512) {
    float ss = v[0]*v[0] + v[1]*v[1] + v[2]*v[2] + v[3]*v[3];
#pragma unroll
    for (int d = 1; d < 32; d <<= 1) ss += __shfl_xor(ss, d, 32);
    if ((t & 31) == 0) red[t >> 5] = ss;
    __syncthreads();
    if (t == 0) {
      float s = 0.0f;
#pragma unroll
      for (int j = 0; j < 8; ++j) s += red[j];
      stot = fmaxf(sqrtf(s), 1e-12f);
    }
    __syncthreads();
    float inv = 1.0f / stot;
    unsigned short* dst = ((o < 256) ? qnT : knT) + (size_t)b * HWN * CCH;
    int oo = o & 255;
#pragma unroll
    for (int k = 0; k < 4; ++k)
      dst[(size_t)(t + k * 256) * CCH + oo] = f2bf(v[k] * inv);
  } else {
    unsigned short* dst = vb + ((size_t)b * CCH + (o - 512)) * HWN;
#pragma unroll
    for (int k = 0; k < 4; ++k) dst[t + k * 256] = f2bf(v[k]);
  }
}

// ---------------- kernel 5: flash attention + masked accumulate ----------------
// grid (i_tiles=16, heads=4, B=8), block 128 (4 waves, 16 queries per wave)
__global__ void la_attn_kernel(const unsigned short* __restrict__ qnT,
                               const unsigned short* __restrict__ knT,
                               const unsigned short* __restrict__ vb,
                               const float* __restrict__ mask,
                               float* __restrict__ out) {
  __shared__ unsigned short pLds[4][16 * 32];  // per-wave P tile [i][j] bf16
  int b = blockIdx.z, h = blockIdx.y, i0 = blockIdx.x * 64;
  int tid = threadIdx.x;
  int wave = tid >> 5, lane = tid & 31;
  int lm = lane & 15, hi = (lane >> 4) & 1;
  int iw = i0 + wave * 16;
  int hc = h * HD;

  const unsigned short* qb = qnT + (size_t)b * HWN * CCH;
  const unsigned short* kb = knT + (size_t)b * HWN * CCH;
  const unsigned short* vp = vb  + (size_t)b * CCH * HWN;

  // Q A-fragments (K = c, 0..31 and 32..63)
  Frag A0, A1;
  {
    const unsigned short* qrow = qb + (size_t)(iw + lm) * CCH + hc;
    int kbA = hi * 8;
    A0.q[0] = *(const v4u*)(qrow + kbA);
    A0.q[1] = *(const v4u*)(qrow + 16 + kbA);
    A1.q[0] = *(const v4u*)(qrow + 32 + kbA);
    A1.q[1] = *(const v4u*)(qrow + 48 + kbA);
  }
  float mrow[8], lrow[8];
#pragma unroll
  for (int r = 0; r < 8; ++r) { mrow[r] = -3.0e38f; lrow[r] = 0.0f; }
  v8f O[4] = {};
  const float scale = 0.125f;   // 1/sqrt(64)
  unsigned short* pl = &pLds[wave][0];
  int kk = hi * 8;              // B-fragment K sub-offset per lane half
  int kbA = hi * 8;             // A-fragment K sub-offset per lane half

  for (int j0 = 0; j0 < HWN; j0 += 32) {
    // ---- S = (Qn)^T Kn over c=64, two j sub-tiles ----
    v8f S[2];
#pragma unroll
    for (int jt = 0; jt < 2; ++jt) {
      const unsigned short* krow = kb + (size_t)(j0 + jt * 16 + lm) * CCH + hc;
      Frag B0, B1;
      B0.q[0] = *(const v4u*)(krow + kk);
      B0.q[1] = *(const v4u*)(krow + kk + 16);
      B1.q[0] = *(const v4u*)(krow + 32 + kk);
      B1.q[1] = *(const v4u*)(krow + 32 + kk + 16);
      v8f s = {};
      s = wmma_bf16(A0, B0, s);
      s = wmma_bf16(A1, B1, s);
      S[jt] = s;
    }
    // ---- online softmax per row (rows live across 16-lane halves) ----
#pragma unroll
    for (int r = 0; r < 8; ++r) {
      float a = S[0][r] * scale, c = S[1][r] * scale;
      float mx = fmaxf(a, c);
#pragma unroll
      for (int d = 1; d < 16; d <<= 1) mx = fmaxf(mx, __shfl_xor(mx, d, 32));
      float mn = fmaxf(mrow[r], mx);
      float cr = __expf(mrow[r] - mn);
      float p0 = __expf(a - mn), p1 = __expf(c - mn);
      float ps = p0 + p1;
#pragma unroll
      for (int d = 1; d < 16; d <<= 1) ps += __shfl_xor(ps, d, 32);
      lrow[r] = lrow[r] * cr + ps;
      mrow[r] = mn;
      int irow = r + 8 * hi;
      pl[irow * 32 + lm]      = f2bf(p0);
      pl[irow * 32 + 16 + lm] = f2bf(p1);
#pragma unroll
      for (int t = 0; t < 4; ++t) O[t][r] *= cr;
    }
    __syncthreads();
    // ---- P D-layout -> A-layout via LDS ----
    Frag P;
    P.q[0] = *(const v4u*)(pl + lm * 32 + kbA);
    P.q[1] = *(const v4u*)(pl + lm * 32 + 16 + kbA);
    // ---- O += P * V ----
#pragma unroll
    for (int t = 0; t < 4; ++t) {
      const unsigned short* vrow = vp + (size_t)(hc + t * 16 + lm) * HWN + j0 + kk;
      Frag Bv;
      Bv.q[0] = *(const v4u*)(vrow);
      Bv.q[1] = *(const v4u*)(vrow + 16);
      O[t] = wmma_bf16(P, Bv, O[t]);
    }
    __syncthreads();
  }
  // ---- epilogue: out[b][hc+c][i] += (O/l) * mask[b][i] ----
#pragma unroll
  for (int r = 0; r < 8; ++r) {
    int irow = iw + r + 8 * hi;
    float f = (1.0f / fmaxf(lrow[r], 1e-30f)) * mask[(size_t)b * HWN + irow];
#pragma unroll
    for (int t = 0; t < 4; ++t) {
      size_t idx = ((size_t)b * CCH + hc + t * 16 + lm) * HWN + irow;
      out[idx] += O[t][r] * f;
    }
  }
}

// ---------------- host launcher ----------------
extern "C" void kernel_launch(void* const* d_in, const int* in_sizes, int n_in,
                              void* d_out, int out_size, void* d_ws, size_t ws_size,
                              hipStream_t stream) {
  const float* x      = (const float*)d_in[0];
  const float* w_inst = (const float*)d_in[1];
  const float* b_inst = (const float*)d_in[2];
  const float* w_q    = (const float*)d_in[3];
  const float* b_q    = (const float*)d_in[4];
  const float* w_k    = (const float*)d_in[5];
  const float* b_k    = (const float*)d_in[6];
  const float* w_v    = (const float*)d_in[7];
  const float* b_v    = (const float*)d_in[8];
  float* out = (float*)d_out;
  char*  ws  = (char*)d_ws;

  // workspace layout (bytes)
  float*          masks   = (float*)(ws);                               // 8*8192*4   = 256 KB
  unsigned short* wqkv    = (unsigned short*)(ws + 0x40000);            // 768*256*2  = 384 KB
  float*          biasqkv = (float*)(ws + 0xA0000);                     // 3 KB
  float*          qkv     = (float*)(ws + 0xA1000);                     // 8*768*1024*4 = 24 MB
  unsigned short* qnT     = (unsigned short*)(ws + 0xA1000 + 25165824); // 4 MB
  unsigned short* knT     = qnT + (size_t)BATCH * HWN * CCH;            // 4 MB
  unsigned short* vbuf    = knT + (size_t)BATCH * HWN * CCH;            // 4 MB

  la_zero_kernel<<<8192, 256, 0, stream>>>(out, BATCH * CCH * HWN);
  la_prep_kernel<<<OQKV, 256, 0, stream>>>(w_q, w_k, w_v, b_q, b_k, b_v, wqkv, biasqkv);
  la_mask_kernel<<<32, 256, 0, stream>>>(x, w_inst, b_inst, masks);

  for (int inst = 0; inst < NI; ++inst) {
    const float* m = masks + (size_t)inst * 8192;
    la_qkv_kernel<<<dim3(16, 12, 8), 128, 0, stream>>>(x, m, wqkv, biasqkv, qkv);
    la_norm_kernel<<<BATCH * OQKV, 256, 0, stream>>>(qkv, qnT, knT, vbuf);
    la_attn_kernel<<<dim3(16, 4, 8), 128, 0, stream>>>(qnT, knT, vbuf, m, out);
  }
}
